// Mamba3DExtra_35991825940957
// MI455X (gfx1250) — matrix operand order repaired and checked by reading, not compiled, not run on hardware
//
#include <hip/hip_runtime.h>
#include <hip/hip_bf16.h>

// ---------------------------------------------------------------------------
// Mamba3D tri-axis bidirectional scan for MI455X (gfx1250, wave32, WMMA).
// One workgroup (256 threads = 8 wave32) owns one row (32 tokens x 256 dims),
// runs the full fwd+bwd mamba block fused in LDS, GEMMs via
// v_wmma_f32_16x16x32_bf16 (bf16 in, f32 accumulate).
// ---------------------------------------------------------------------------

typedef __attribute__((ext_vector_type(16))) __bf16 v16bf;
typedef __attribute__((ext_vector_type(8)))  float  v8f;

#define LTOK   32
#define DMODEL 256
#define DINNER 512
#define DSTATE 16
#define DTRANK 16

// LDS layout (bytes), total ~215 KB of the 320 KB/WGP
#define OFF_XRES   0                        // 32*256  f32  = 32768  residual row
#define OFF_XN     (OFF_XRES + 32768)       // 32*256  bf16 = 16384  normed (GEMM1 A)
#define OFF_XZ     (OFF_XN   + 16384)       // 32*1024 bf16 = 65536  [xi | z]; xi cols reused for y
#define OFF_XC     (OFF_XZ   + 65536)       // 32*512  bf16 = 32768  conv+silu out
#define OFF_DT     (OFF_XC   + 32768)       // 32*512  f32  = 65536  softplus(dt)
#define OFF_DBL    (OFF_DT   + 65536)       // 32*48   f32  = 6144   [dt_r | B | C]
#define OFF_RED    (OFF_DBL  + 6144)        // 32*8    f32  = 1024   rmsnorm scratch
#define SMEM_BYTES (OFF_RED  + 1024)

struct BlockP {
    const __bf16* inT;     // [1024][256]  in_w transposed, bf16
    const __bf16* xprojT;  // [48][512]    xproj_w transposed, bf16
    const __bf16* outT;    // [256][512]   out_w transposed, bf16
    const float*  A;       // [512][16]    -exp(A_log)
    const float*  norm_w;  // [256]
    const float*  conv_w;  // [512][4]
    const float*  conv_b;  // [512]
    const float*  dt_w;    // [16][512]
    const float*  dt_b;    // [512]
    const float*  Dp;      // [512]
};

// A fragment (16x32 bf16, row-major in LDS). Lanes 0-15: row m, K = kb+[0..15];
// lanes 16-31: row m, K = kb+[16..31]  (contiguous order, matches load_Bt).
__device__ inline v16bf load_A16(const __bf16* base, int ld, int m0, int kb, int lane) {
    int m  = m0 + (lane & 15);
    int ko = kb + ((lane & 16) ? 16 : 0);
    const __bf16* p = base + m * ld + ko;
    v16bf a;
    ((int4*)&a)[0] = *reinterpret_cast<const int4*>(p);       // K ko..ko+7
    ((int4*)&a)[1] = *reinterpret_cast<const int4*>(p + 8);   // K ko+8..ko+15
    return a;
}

// B fragment (32x16 bf16) from column-major (N-major) weights WT[n][k].
// Lanes 0-15 hold K = kb..kb+15 for column n=lane; lanes 16-31 hold K = kb+16..kb+31.
__device__ inline v16bf load_Bt(const __bf16* colptr, int kb) {
    v16bf b;
    ((int4*)&b)[0] = *reinterpret_cast<const int4*>(colptr + kb);
    ((int4*)&b)[1] = *reinterpret_cast<const int4*>(colptr + kb + 8);
    return b;
}

__device__ inline float silu_f(float v) { return v / (1.f + __expf(-v)); }

__device__ void mamba_block(char* smem, const BlockP p, int rev) {
    const int tid  = threadIdx.x;
    const int lane = tid & 31;
    const int wave = tid >> 5;
    const int l15  = lane & 15;
    const int hi   = (lane & 16) ? 1 : 0;

    float*  xres = (float*)(smem + OFF_XRES);
    __bf16* xn   = (__bf16*)(smem + OFF_XN);
    __bf16* xz   = (__bf16*)(smem + OFF_XZ);   // 32 x 1024: cols 0..511 = xi (later y), 512..1023 = z
    __bf16* xc   = (__bf16*)(smem + OFF_XC);
    float*  dt   = (float*)(smem + OFF_DT);
    float*  dbl  = (float*)(smem + OFF_DBL);
    float*  red  = (float*)(smem + OFF_RED);

    // Warm L2/WGP$ for the big weight streams while we do RMSNorm.
    __builtin_prefetch(p.inT  + (size_t)tid * 1024, 0, 0);
    __builtin_prefetch(p.outT + (size_t)tid * 512, 0, 0);

    // ---- 1. RMSNorm -> xn (bf16) : 8 threads per token, 32 elems each ----
    {
        int t = tid >> 3, g = tid & 7;
        const float* xr = xres + t * DMODEL + g * 32;
        float ss = 0.f;
        #pragma unroll
        for (int i = 0; i < 32; ++i) { float v = xr[i]; ss += v * v; }
        red[t * 8 + g] = ss;
        __syncthreads();
        if (g == 0) {
            float s = 0.f;
            #pragma unroll
            for (int i = 0; i < 8; ++i) s += red[t * 8 + i];
            red[t * 8] = rsqrtf(s * (1.f / 256.f) + 1e-5f);
        }
        __syncthreads();
        float sc = red[t * 8];
        #pragma unroll
        for (int i = 0; i < 32; ++i) {
            int e = g * 32 + i;
            xn[t * DMODEL + e] = (__bf16)(xres[t * DMODEL + e] * sc * p.norm_w[e]);
        }
    }
    __syncthreads();

    // ---- 2. in_proj GEMM: (32x256) @ (256x1024) -> xz  (WMMA bf16) ----
    // 2 M-tiles x 64 N-tiles; wave w owns N-tiles [8w, 8w+8).
    // Both M-tiles share each B fragment (halves global weight traffic).
    for (int nt = wave * 8; nt < wave * 8 + 8; ++nt) {
        const __bf16* Bp = p.inT + (size_t)(nt * 16 + l15) * DMODEL + hi * 16;
        v8f acc0 = {}, acc1 = {};
        #pragma unroll
        for (int kb = 0; kb < DMODEL; kb += 32) {
            v16bf b  = load_Bt(Bp, kb);
            v16bf a0 = load_A16(xn, DMODEL, 0,  kb, lane);
            v16bf a1 = load_A16(xn, DMODEL, 16, kb, lane);
            acc0 = __builtin_amdgcn_wmma_f32_16x16x32_bf16(
                false, a0, false, b, (short)0, acc0, false, false);
            acc1 = __builtin_amdgcn_wmma_f32_16x16x32_bf16(
                false, a1, false, b, (short)0, acc1, false, false);
        }
        int col = nt * 16 + l15;        // 0..1023 in merged xi|z buffer
        int rb  = hi * 8;
        #pragma unroll
        for (int r = 0; r < 8; ++r) {
            xz[(rb + r)      * 1024 + col] = (__bf16)acc0[r];
            xz[(rb + r + 16) * 1024 + col] = (__bf16)acc1[r];
        }
    }
    __syncthreads();

    // ---- 3. depthwise causal conv (width 4) + SiLU -> xc ----
    {
        int d0 = tid * 2;
        #pragma unroll 1
        for (int t = 0; t < LTOK; ++t) {
            #pragma unroll
            for (int c = 0; c < 2; ++c) {
                int d = d0 + c;
                float acc = p.conv_b[d];
                #pragma unroll
                for (int j = 0; j < 4; ++j) {
                    int tt = rev ? (t + 3 - j) : (t - 3 + j);
                    if (tt >= 0 && tt < LTOK)
                        acc += p.conv_w[d * 4 + j] * (float)xz[tt * 1024 + d];
                }
                xc[t * DINNER + d] = (__bf16)silu_f(acc);
            }
        }
    }
    __syncthreads();

    // ---- 4. x_proj GEMM: (32x512) @ (512x48) -> dbl (WMMA, 6 tiles on 6 waves) ----
    if (wave < 6) {
        int mt = wave & 1, nt = wave >> 1;
        const __bf16* Bp = p.xprojT + (size_t)(nt * 16 + l15) * DINNER + hi * 16;
        v8f acc = {};
        #pragma unroll 4
        for (int kb = 0; kb < DINNER; kb += 32) {
            v16bf a = load_A16(xc, DINNER, mt * 16, kb, lane);
            v16bf b = load_Bt(Bp, kb);
            acc = __builtin_amdgcn_wmma_f32_16x16x32_bf16(
                false, a, false, b, (short)0, acc, false, false);
        }
        int col = nt * 16 + l15, rb = mt * 16 + hi * 8;
        #pragma unroll
        for (int r = 0; r < 8; ++r) dbl[(rb + r) * 48 + col] = acc[r];
    }
    __syncthreads();

    // ---- 5. dt_proj (K=16, VALU) + softplus -> dt ----
    {
        #pragma unroll 1
        for (int t = 0; t < LTOK; ++t) {
            #pragma unroll
            for (int c = 0; c < 2; ++c) {
                int d = tid * 2 + c;
                float a = p.dt_b[d];
                #pragma unroll
                for (int r = 0; r < DTRANK; ++r)
                    a += dbl[t * 48 + r] * p.dt_w[r * DINNER + d];
                dt[t * DINNER + d] = (a > 20.f) ? a : log1pf(__expf(a));
            }
        }
    }
    __syncthreads();

    // ---- 6. selective scan (2 channels/thread, state 16 each) + D + gate -> y ----
    {
        int d0 = tid * 2, d1 = d0 + 1;
        float A0[DSTATE], A1[DSTATE], h0[DSTATE], h1[DSTATE];
        #pragma unroll
        for (int n = 0; n < DSTATE; ++n) {
            A0[n] = p.A[d0 * DSTATE + n];
            A1[n] = p.A[d1 * DSTATE + n];
            h0[n] = 0.f; h1[n] = 0.f;
        }
        float D0 = p.Dp[d0], D1 = p.Dp[d1];
        #pragma unroll 1
        for (int s = 0; s < LTOK; ++s) {
            int t = rev ? (LTOK - 1 - s) : s;
            const float* bc = dbl + t * 48;
            float dt0 = dt[t * DINNER + d0], dt1 = dt[t * DINNER + d1];
            float x0 = (float)xc[t * DINNER + d0], x1 = (float)xc[t * DINNER + d1];
            float u0 = dt0 * x0, u1 = dt1 * x1;
            float y0 = 0.f, y1 = 0.f;
            #pragma unroll
            for (int n = 0; n < DSTATE; ++n) {
                float Bn = bc[16 + n], Cn = bc[32 + n];
                h0[n] = h0[n] * __expf(dt0 * A0[n]) + u0 * Bn;
                h1[n] = h1[n] * __expf(dt1 * A1[n]) + u1 * Bn;
                y0 += h0[n] * Cn;
                y1 += h1[n] * Cn;
            }
            y0 += x0 * D0;  y1 += x1 * D1;
            float z0 = (float)xz[t * 1024 + 512 + d0], z1 = (float)xz[t * 1024 + 512 + d1];
            y0 *= silu_f(z0);
            y1 *= silu_f(z1);
            xz[t * 1024 + d0] = (__bf16)y0;   // overwrite xi columns with gated y
            xz[t * 1024 + d1] = (__bf16)y1;
        }
    }
    __syncthreads();

    // ---- 7. out_proj GEMM: (32x512) @ (512x256) + residual -> xres (WMMA) ----
    // y lives in xz cols 0..511 with ld=1024. B fragment shared by both M-tiles.
    {
        #pragma unroll
        for (int q = 0; q < 2; ++q) {
            int nt = wave * 2 + q;
            const __bf16* Bp = p.outT + (size_t)(nt * 16 + l15) * DINNER + hi * 16;
            v8f acc0 = {}, acc1 = {};
            #pragma unroll 4
            for (int kb = 0; kb < DINNER; kb += 32) {
                v16bf b  = load_Bt(Bp, kb);
                v16bf a0 = load_A16(xz, 1024, 0,  kb, lane);
                v16bf a1 = load_A16(xz, 1024, 16, kb, lane);
                acc0 = __builtin_amdgcn_wmma_f32_16x16x32_bf16(
                    false, a0, false, b, (short)0, acc0, false, false);
                acc1 = __builtin_amdgcn_wmma_f32_16x16x32_bf16(
                    false, a1, false, b, (short)0, acc1, false, false);
            }
            int col = nt * 16 + l15, rb = hi * 8;
            #pragma unroll
            for (int r = 0; r < 8; ++r) {
                xres[(rb + r)      * DMODEL + col] += acc0[r];
                xres[(rb + r + 16) * DMODEL + col] += acc1[r];
            }
        }
    }
    __syncthreads();
}

// One workgroup per row; axis selects stride (the "transposes" are free).
__global__ __launch_bounds__(256, 1)
void mamba_axis_kernel(float* __restrict__ data, const unsigned char* __restrict__ mask,
                       BlockP fwd, BlockP bwd, int axis) {
    extern __shared__ char smem[];
    int row = blockIdx.x;
    if (!mask[row]) return;   // masked-off rows stay as-is (in-place update)

    int n = row >> 10, rem = row & 1023, i = rem >> 5, j = rem & 31;
    long base, strideT;
    if (axis == 0)      { base = ((long)n * 32768 + i * 32   + j     ) * 256; strideT = 32 * 32 * 256; } // seq = L
    else if (axis == 1) { base = ((long)n * 32768 + i * 1024 + j     ) * 256; strideT = 32 * 256;      } // seq = H
    else                { base = ((long)n * 32768 + i * 1024 + j * 32) * 256; strideT = 256;           } // seq = W

    float* xres = (float*)(smem + OFF_XRES);
    {
        int t = threadIdx.x >> 3, g = threadIdx.x & 7;
        const float4* src = (const float4*)(data + base + (long)t * strideT + g * 32);
        float4* dst = (float4*)(xres + t * DMODEL + g * 32);
        #pragma unroll
        for (int q = 0; q < 8; ++q) dst[q] = src[q];
    }
    __syncthreads();

    mamba_block(smem, fwd, 0);
    mamba_block(smem, bwd, 1);

    {
        int t = threadIdx.x >> 3, g = threadIdx.x & 7;
        float4* dst = (float4*)(data + base + (long)t * strideT + g * 32);
        const float4* src = (const float4*)(xres + t * DMODEL + g * 32);
        #pragma unroll
        for (int q = 0; q < 8; ++q) dst[q] = src[q];
    }
}

// Pre-pass: fp32 -> bf16 weight transposes (WMMA B wants N-major) + A = -exp(A_log).
__global__ void convert_params_kernel(const float* __restrict__ in_w,
                                      const float* __restrict__ xproj_w,
                                      const float* __restrict__ out_w,
                                      const float* __restrict__ A_log,
                                      __bf16* __restrict__ inT,
                                      __bf16* __restrict__ xprojT,
                                      __bf16* __restrict__ outT,
                                      float* __restrict__ Aout) {
    int i = blockIdx.x * blockDim.x + threadIdx.x;
    int stride = gridDim.x * blockDim.x;
    for (int idx = i; idx < 1024 * 256; idx += stride) {       // inT[n][k] = in_w[k][n]
        int nn = idx >> 8, k = idx & 255;
        inT[idx] = (__bf16)in_w[k * 1024 + nn];
    }
    for (int idx = i; idx < 48 * 512; idx += stride) {         // xprojT[n][k]
        int nn = idx >> 9, k = idx & 511;
        xprojT[idx] = (__bf16)xproj_w[k * 48 + nn];
    }
    for (int idx = i; idx < 256 * 512; idx += stride) {        // outT[n][k]
        int nn = idx >> 9, k = idx & 511;
        outT[idx] = (__bf16)out_w[k * 256 + nn];
    }
    for (int idx = i; idx < 512 * 16; idx += stride)           // A = -exp(A_log)
        Aout[idx] = -__expf(A_log[idx]);
}

extern "C" void kernel_launch(void* const* d_in, const int* in_sizes, int n_in,
                              void* d_out, int out_size, void* d_ws, size_t ws_size,
                              hipStream_t stream) {
    (void)in_sizes; (void)n_in; (void)ws_size;
    const float* x = (const float*)d_in[0];
    const unsigned char* masks[3] = { (const unsigned char*)d_in[1],
                                      (const unsigned char*)d_in[2],
                                      (const unsigned char*)d_in[3] };
    // Params: 6 blocks x 10 arrays starting at d_in[4]:
    // 0 norm_w, 1 in_w, 2 conv_w, 3 conv_b, 4 xproj_w, 5 dtproj_w,
    // 6 dtproj_b, 7 A_log, 8 Dp, 9 out_w
    // Workspace layout (bytes)
    char* ws = (char*)d_ws;
    const size_t SZ_INT   = 1024 * 256 * sizeof(__bf16);   // 512 KB
    const size_t SZ_XPT   = 48 * 512 * sizeof(__bf16);     // 48 KB
    const size_t SZ_OUTT  = 256 * 512 * sizeof(__bf16);    // 256 KB
    const size_t SZ_A     = 512 * 16 * sizeof(float);      // 32 KB
    __bf16* inT_base    = (__bf16*)ws;
    __bf16* xprojT_base = (__bf16*)(ws + 6 * SZ_INT);
    __bf16* outT_base   = (__bf16*)(ws + 6 * SZ_INT + 6 * SZ_XPT);
    float*  A_base      = (float*) (ws + 6 * SZ_INT + 6 * SZ_XPT + 6 * SZ_OUTT);

    BlockP bp[6];
    for (int b = 0; b < 6; ++b) {
        const float* norm_w = (const float*)d_in[4 + b * 10 + 0];
        const float* in_w   = (const float*)d_in[4 + b * 10 + 1];
        const float* conv_w = (const float*)d_in[4 + b * 10 + 2];
        const float* conv_b = (const float*)d_in[4 + b * 10 + 3];
        const float* xproj  = (const float*)d_in[4 + b * 10 + 4];
        const float* dt_w   = (const float*)d_in[4 + b * 10 + 5];
        const float* dt_b   = (const float*)d_in[4 + b * 10 + 6];
        const float* A_log  = (const float*)d_in[4 + b * 10 + 7];
        const float* Dp     = (const float*)d_in[4 + b * 10 + 8];
        const float* out_w  = (const float*)d_in[4 + b * 10 + 9];

        __bf16* inT    = inT_base    + b * (SZ_INT  / sizeof(__bf16));
        __bf16* xprojT = xprojT_base + b * (SZ_XPT  / sizeof(__bf16));
        __bf16* outT   = outT_base   + b * (SZ_OUTT / sizeof(__bf16));
        float*  Aout   = A_base      + b * (SZ_A    / sizeof(float));

        convert_params_kernel<<<512, 256, 0, stream>>>(in_w, xproj, out_w, A_log,
                                                       inT, xprojT, outT, Aout);
        bp[b].inT = inT;   bp[b].xprojT = xprojT; bp[b].outT = outT; bp[b].A = Aout;
        bp[b].norm_w = norm_w; bp[b].conv_w = conv_w; bp[b].conv_b = conv_b;
        bp[b].dt_w = dt_w; bp[b].dt_b = dt_b; bp[b].Dp = Dp;
    }

    // Seed d_out with the input volume; all three passes then update it in place.
    hipMemcpyAsync(d_out, x, (size_t)out_size * sizeof(float),
                   hipMemcpyDeviceToDevice, stream);

    (void)hipFuncSetAttribute((const void*)mamba_axis_kernel,
                              hipFuncAttributeMaxDynamicSharedMemorySize, SMEM_BYTES);

    float* vol = (float*)d_out;
    // axis 0: rows along L (params l / rl), axis 1: along H (h / rh), axis 2: along W (w / rw)
    mamba_axis_kernel<<<4096, 256, SMEM_BYTES, stream>>>(vol, masks[0], bp[0], bp[1], 0);
    mamba_axis_kernel<<<4096, 256, SMEM_BYTES, stream>>>(vol, masks[1], bp[2], bp[3], 1);
    mamba_axis_kernel<<<4096, 256, SMEM_BYTES, stream>>>(vol, masks[2], bp[4], bp[5], 2);
}
